// Mamba_45844480917622
// MI455X (gfx1250) — matrix operand order repaired and checked by reading, not compiled, not run on hardware
//
#include <hip/hip_runtime.h>

#define HIDN 256
#define EXPN 512
#define NLAY 4
#define HISTN 100
#define SEQN 2048
#define NBAT 32
#define MTOT (NBAT * SEQN)   // 65536 token rows

typedef _Float16 h16;
typedef __attribute__((ext_vector_type(16))) _Float16 v16h;
typedef __attribute__((ext_vector_type(8)))  _Float16 v8h;
typedef __attribute__((ext_vector_type(8)))  float    v8f;

enum { EP_BIAS = 0, EP_GATE = 1, EP_SCAN = 2, EP_RESID = 3 };

// async DMA: global memory -> LDS, 16 bytes per lane, tracked by ASYNCcnt.
// lds = LDS byte address (low 32 bits of the generic shared pointer),
// gaddr = 64-bit global address.
__device__ __forceinline__ void async_copy_b128(unsigned lds, unsigned long long gaddr) {
  asm volatile("global_load_async_to_lds_b128 %0, %1, off"
               :: "v"(lds), "v"(gaddr) : "memory");
}
__device__ __forceinline__ void wait_async0() {
  asm volatile("s_wait_asynccnt 0x0" ::: "memory");
}

// ---------------------------------------------------------------------------
// Tiled WMMA GEMM: C(MxN) = A(MxK,f16) * B(KxN) with B pre-transposed to
// Bt(NxK,f16).  Block tile 128x64, K-step 32, 256 threads = 8 waves arranged
// 4(M) x 2(N); each wave owns a 32x32 strip -> 2x2 accums of 16x16 f32,
// i.e. 4 WMMAs per K-step amortizing 2 A-frags + 2 B-frags.
// Double-buffered LDS fed by GLOBAL_LOAD_ASYNC_TO_LDS_B128 (no VGPR round
// trip); one s_wait_asynccnt + barrier per K-step.
// Fragment layouts per CDNA5 ISA 7.12.2:
//   A 16x32 f16 : lane = M row (lanes 0-15 / 16-31 repeat rows), VGPR0-3 hold
//                 K = kb..kb+7 and VGPR4-7 hold K = kb+16..kb+23, kb = hi*8.
//   B 32x16 f16 : lane = N col, same K split as A: elems 0-7 K = kb..kb+7, elems 8-15 K = kb+16..kb+23.
// ---------------------------------------------------------------------------
template <int MODE>
__global__ __launch_bounds__(256) void gemm_wmma(
    const h16* __restrict__ A, const h16* __restrict__ Bt,
    const float* __restrict__ bias,
    float* __restrict__ Cf, h16* __restrict__ Ch,
    const float* __restrict__ e0, const float* __restrict__ e1,
    const float* __restrict__ e2,
    int M, int N, int K, int strideAz, int strideBz, int rowStrideZ)
{
  __shared__ __attribute__((aligned(16))) h16 As[2][128][40];
  __shared__ __attribute__((aligned(16))) h16 Bs[2][64][40];

  const int tid  = threadIdx.x;
  const int lane = tid & 31;
  const int wid  = tid >> 5;
  const int wm   = wid & 3;      // 0..3 : M strip (32 rows each)
  const int wn   = wid >> 2;     // 0..1 : N strip (32 cols each)
  const int lrow = lane & 15;
  const int hi   = lane >> 4;
  const int kbA  = hi * 8;
  const int kbB  = hi * 16;

  const int rowBase = blockIdx.y * 128;
  const int colBase = blockIdx.x * 64;
  const int z = blockIdx.z;
  const h16* Ab = A  + (size_t)z * strideAz;
  const h16* Bb = Bt + (size_t)z * strideBz;

  // loader mapping: A tile 128x32 (16 halves/thread), B tile 64x32 (8/thread)
  const int ldrA = tid >> 1;          // 0..127
  const int ldcA = (tid & 1) * 16;    // 0,16
  const int ldrB = tid >> 2;          // 0..63
  const int ldcB = (tid & 3) * 8;     // 0,8,16,24

  const h16* Aptr = Ab + (size_t)(rowBase + ldrA) * K + ldcA;
  const h16* Bptr = Bb + (size_t)(colBase + ldrB) * K + ldcB;

  // per-lane LDS destinations for the async DMA (byte addresses)
  unsigned ldsA[2], ldsB[2];
  ldsA[0] = (unsigned)(uintptr_t)&As[0][ldrA][ldcA];
  ldsA[1] = (unsigned)(uintptr_t)&As[1][ldrA][ldcA];
  ldsB[0] = (unsigned)(uintptr_t)&Bs[0][ldrB][ldcB];
  ldsB[1] = (unsigned)(uintptr_t)&Bs[1][ldrB][ldcB];

  // prologue: stage K-tile 0 into buffer 0 via async DMA
  async_copy_b128(ldsA[0],      (unsigned long long)(uintptr_t)Aptr);
  async_copy_b128(ldsA[0] + 16, (unsigned long long)(uintptr_t)(Aptr + 8));
  async_copy_b128(ldsB[0],      (unsigned long long)(uintptr_t)Bptr);
  wait_async0();
  __syncthreads();

  v8f acc[2][2] = {};

  const int nk = K / 32;
  for (int ki = 0; ki < nk; ++ki) {
    const int cur = ki & 1;

    // kick off async DMA of the next K-tile into the other buffer
    // (that buffer's last readers finished before the previous barrier)
    const bool hasNext = (ki + 1 < nk);
    if (hasNext) {
      const int nxt = cur ^ 1;
      const h16* Ap = Aptr + (size_t)(ki + 1) * 32;
      const h16* Bp = Bptr + (size_t)(ki + 1) * 32;
      async_copy_b128(ldsA[nxt],      (unsigned long long)(uintptr_t)Ap);
      async_copy_b128(ldsA[nxt] + 16, (unsigned long long)(uintptr_t)(Ap + 8));
      async_copy_b128(ldsB[nxt],      (unsigned long long)(uintptr_t)Bp);
    }
    if (ki + 2 < nk)   // pull tile i+2 toward near caches (global_prefetch_b8)
      __builtin_prefetch(Aptr + (size_t)(ki + 2) * 32, 0, 3);

    // fragments from current buffer + 4 WMMAs
    v16h af0, af1, bf0, bf1;
    const h16* ap0 = &As[cur][wm * 32 + lrow][0];
    const h16* ap1 = &As[cur][wm * 32 + 16 + lrow][0];
    const h16* bp0 = &Bs[cur][wn * 32 + lrow][0];
    const h16* bp1 = &Bs[cur][wn * 32 + 16 + lrow][0];
#pragma unroll
    for (int i = 0; i < 8; ++i) {
      af0[i]     = ap0[kbA + i];
      af0[i + 8] = ap0[kbA + 16 + i];
      af1[i]     = ap1[kbA + i];
      af1[i + 8] = ap1[kbA + 16 + i];
    }
#pragma unroll
    for (int i = 0; i < 8; ++i) {
      bf0[i]     = bp0[kbA + i];
      bf0[i + 8] = bp0[kbA + 16 + i];
      bf1[i]     = bp1[kbA + i];
      bf1[i + 8] = bp1[kbA + 16 + i];
    }
    acc[0][0] = __builtin_amdgcn_wmma_f32_16x16x32_f16(false, af0, false, bf0,
                                                       (short)0, acc[0][0], false, false);
    acc[0][1] = __builtin_amdgcn_wmma_f32_16x16x32_f16(false, af0, false, bf1,
                                                       (short)0, acc[0][1], false, false);
    acc[1][0] = __builtin_amdgcn_wmma_f32_16x16x32_f16(false, af1, false, bf0,
                                                       (short)0, acc[1][0], false, false);
    acc[1][1] = __builtin_amdgcn_wmma_f32_16x16x32_f16(false, af1, false, bf1,
                                                       (short)0, acc[1][1], false, false);

    // publish the async-filled buffer for the next iteration
    wait_async0();
    __syncthreads();
  }

  // D layout: lane&15 = N col, lane>>4 picks M half, VGPR r -> M = hi*8 + r.
  // Index math kept incremental (one mul per column, adds in the row loop).
#pragma unroll
  for (int si = 0; si < 2; ++si) {
    const int mBase = rowBase + wm * 32 + si * 16 + hi * 8;
    const size_t growBase = (size_t)z * rowStrideZ + mBase;
#pragma unroll
    for (int j = 0; j < 2; ++j) {
      const int n = colBase + wn * 32 + j * 16 + lrow;
      const float bn = bias[n];
      size_t idx  = growBase * N + n;
      size_t idxz = growBase * (2 * EXPN) + EXPN + n;   // EP_SCAN z-gate stream
#pragma unroll
      for (int r = 0; r < 8; ++r) {
        const float v = acc[si][j][r];
        if (MODE == EP_BIAS) {
          Cf[idx] = v + bn;
        } else if (MODE == EP_GATE) {
          float t  = v + bn;
          float sp = (t > 20.0f) ? t : log1pf(__expf(t));
          Cf[idx] = 1.0f / (1.0f + __expf(-sp));
        } else if (MODE == EP_SCAN) {
          float g  = e0[idx];                       // sigmoid(softplus(dt))
          float xv = e1[idx];                       // xc (f32)
          float zv = e2[idxz];                      // z half of xz
          float y  = v * g + xv * bn;               // bn = D[n]
          y *= zv / (1.0f + __expf(-zv));           // * silu(z)
          Ch[idx] = (h16)y;
        } else { // EP_RESID (in-place residual on h)
          Cf[idx] = v + bn + Cf[idx];
        }
        idx  += N;
        idxz += 2 * EXPN;
      }
    }
  }
}

// ---------------------------------------------------------------------------
// Wave-per-row layernorm over HIDN=256 cols; emits f16 (GEMM operand) or f32.
// ---------------------------------------------------------------------------
template <typename OT>
__global__ __launch_bounds__(256) void k_layernorm(
    const float* __restrict__ x, const float* __restrict__ w,
    const float* __restrict__ b, OT* __restrict__ out, int nrows)
{
  const int lane = threadIdx.x & 31;
  const int wid  = threadIdx.x >> 5;
  const int row  = blockIdx.x * 8 + wid;
  if (row >= nrows) return;
  const float* xr = x + (size_t)row * HIDN;
  float v[8], s = 0.0f, ss = 0.0f;
#pragma unroll
  for (int i = 0; i < 8; ++i) {
    v[i] = xr[lane + i * 32];
    s += v[i]; ss += v[i] * v[i];
  }
#pragma unroll
  for (int o = 16; o > 0; o >>= 1) {
    s  += __shfl_xor(s, o, 32);
    ss += __shfl_xor(ss, o, 32);
  }
  const float mean = s * (1.0f / HIDN);
  const float var  = ss * (1.0f / HIDN) - mean * mean;
  const float rstd = rsqrtf(var + 1e-5f);
  OT* orow = out + (size_t)row * HIDN;
#pragma unroll
  for (int i = 0; i < 8; ++i) {
    const int c = lane + i * 32;
    orow[c] = (OT)((v[i] - mean) * rstd * w[c] + b[c]);
  }
}

// ---------------------------------------------------------------------------
// Depthwise conv (k=3, pad 1) along seq + bias + SiLU.  One thread per (b,e),
// marches s; also writes f16 copy (dt GEMM operand) and the per-batch
// transposed [E][K=128] tile of the first 100 steps (scan GEMM B operand).
// ---------------------------------------------------------------------------
__global__ __launch_bounds__(256) void k_conv_silu(
    const float* __restrict__ xz, const float* __restrict__ convW,
    const float* __restrict__ convB,
    float* __restrict__ xc, h16* __restrict__ xch, h16* __restrict__ xcT)
{
  const int t = blockIdx.x * blockDim.x + threadIdx.x; // [0, NBAT*EXPN)
  const int e = t % EXPN;
  const int b = t / EXPN;
  const float w0 = convW[e * 3 + 0];
  const float w1 = convW[e * 3 + 1];
  const float w2 = convW[e * 3 + 2];
  const float cb = convB[e];
  float xm1 = 0.0f;
  float x0  = xz[(size_t)(b * SEQN) * (2 * EXPN) + e];
  for (int s = 0; s < SEQN; ++s) {
    float xp1 = (s + 1 < SEQN) ? xz[(size_t)(b * SEQN + s + 1) * (2 * EXPN) + e] : 0.0f;
    float v = w0 * xm1 + w1 * x0 + w2 * xp1 + cb;
    v = v / (1.0f + __expf(-v));                 // silu
    const size_t gi = (size_t)(b * SEQN + s) * EXPN + e;
    xc[gi]  = v;
    xch[gi] = (h16)v;
    if (s < HISTN) xcT[((size_t)b * EXPN + e) * 128 + s] = (h16)v;
    xm1 = x0; x0 = xp1;
  }
  for (int k = HISTN; k < 128; ++k)
    xcT[((size_t)b * EXPN + e) * 128 + k] = (h16)0.0f;
}

// Decay matrix W[s][k] = 0.9^(s-k) for k<=s, k<HIST; zero-padded to K=128.
__global__ void k_decay(h16* __restrict__ wdec)
{
  const int idx = blockIdx.x * 256 + threadIdx.x;
  if (idx >= SEQN * 128) return;
  const int s = idx >> 7, k = idx & 127;
  float v = 0.0f;
  if (k < HISTN && k <= s)
    v = __expf((float)(s - k) * -0.105360515657826281f); // ln(0.9)
  wdec[idx] = (h16)v;
}

// f32 [K][N] -> f16 [N][K] transpose (weight prep for GEMM B operands)
__global__ void k_transpose(const float* __restrict__ in, h16* __restrict__ out,
                            int K, int N)
{
  const int idx = blockIdx.x * 256 + threadIdx.x;
  if (idx >= K * N) return;
  const int k = idx / N, n = idx % N;
  out[(size_t)n * K + k] = (h16)in[idx];
}

__global__ void k_cvt(const float* __restrict__ in, h16* __restrict__ out, int n)
{
  const int i = blockIdx.x * 256 + threadIdx.x;
  if (i < n) out[i] = (h16)in[i];
}

// per-batch mean over seq of final-layernormed h
__global__ __launch_bounds__(256) void k_mean(const float* __restrict__ hln,
                                              float* __restrict__ hbar)
{
  const int b = blockIdx.x, j = threadIdx.x;   // 256 threads = HIDN cols
  float s = 0.0f;
  const float* base = hln + (size_t)b * SEQN * HIDN + j;
  for (int i = 0; i < SEQN; ++i) s += base[(size_t)i * HIDN];
  hbar[b * HIDN + j] = s * (1.0f / SEQN);
}

// head: (256->128) + LN + exact gelu + (128->1); one block per batch
__global__ __launch_bounds__(128) void k_head(
    const float* __restrict__ hbar, const float* __restrict__ c1W,
    const float* __restrict__ c1B, const float* __restrict__ clnW,
    const float* __restrict__ clnB, const float* __restrict__ c2W,
    const float* __restrict__ c2B, float* __restrict__ out)
{
  __shared__ float rs[128];
  const int b = blockIdx.x, j = threadIdx.x;
  const float* hb = hbar + b * HIDN;
  float t = c1B[j];
  for (int i = 0; i < HIDN; ++i) t += hb[i] * c1W[i * 128 + j];

  rs[j] = t; __syncthreads();
  for (int o = 64; o > 0; o >>= 1) { if (j < o) rs[j] += rs[j + o]; __syncthreads(); }
  const float mean = rs[0] * (1.0f / 128.0f); __syncthreads();

  rs[j] = (t - mean) * (t - mean); __syncthreads();
  for (int o = 64; o > 0; o >>= 1) { if (j < o) rs[j] += rs[j + o]; __syncthreads(); }
  const float var = rs[0] * (1.0f / 128.0f); __syncthreads();

  const float rstd = rsqrtf(var + 1e-5f);
  float u = (t - mean) * rstd * clnW[j] + clnB[j];
  u = 0.5f * u * (1.0f + erff(u * 0.70710678118654752f)); // exact gelu

  rs[j] = u * c2W[j]; __syncthreads();
  for (int o = 64; o > 0; o >>= 1) { if (j < o) rs[j] += rs[j + o]; __syncthreads(); }
  if (j == 0) out[b] = rs[0] + c2B[0];
}

// ---------------------------------------------------------------------------
extern "C" void kernel_launch(void* const* d_in, const int* in_sizes, int n_in,
                              void* d_out, int out_size, void* d_ws, size_t ws_size,
                              hipStream_t stream) {
  const float* x     = (const float*)d_in[0];
  const float* Wi    = (const float*)d_in[1];
  const float* bi    = (const float*)d_in[2];
  const float* ln_w  = (const float*)d_in[3];
  const float* ln_b  = (const float*)d_in[4];
  const float* inW   = (const float*)d_in[5];
  const float* inB   = (const float*)d_in[6];
  const float* convW = (const float*)d_in[7];
  const float* convB = (const float*)d_in[8];
  const float* dtW   = (const float*)d_in[9];
  const float* dtB   = (const float*)d_in[10];
  const float* Dp    = (const float*)d_in[11];
  const float* outW  = (const float*)d_in[12];
  const float* outB  = (const float*)d_in[13];
  const float* fn_w  = (const float*)d_in[14];
  const float* fn_b  = (const float*)d_in[15];
  const float* c1W   = (const float*)d_in[16];
  const float* c1B   = (const float*)d_in[17];
  const float* clnW  = (const float*)d_in[18];
  const float* clnB  = (const float*)d_in[19];
  const float* c2W   = (const float*)d_in[20];
  const float* c2B   = (const float*)d_in[21];
  float* out = (float*)d_out;

  char* wp = (char*)d_ws;
  auto alloc = [&](size_t bytes) -> char* {
    char* p = wp; wp += (bytes + 255) & ~(size_t)255; return p;
  };
  h16*   xh    = (h16*)  alloc((size_t)MTOT * 128 * 2);          // x as f16
  float* h     = (float*)alloc((size_t)MTOT * HIDN * 4);         // residual stream
  h16*   hn    = (h16*)  alloc((size_t)MTOT * HIDN * 2);         // ln(h) f16
  float* xz    = (float*)alloc((size_t)MTOT * 2 * EXPN * 4);     // in-proj out
  float* xc    = (float*)alloc((size_t)MTOT * EXPN * 4);         // conv+silu f32
  h16*   xch   = (h16*)  alloc((size_t)MTOT * EXPN * 2);         // conv+silu f16
  float* gate  = (float*)alloc((size_t)MTOT * EXPN * 4);         // sig(softplus(dt))
  h16*   yv    = (h16*)  alloc((size_t)MTOT * EXPN * 2);         // scan output f16
  h16*   xcT   = (h16*)  alloc((size_t)NBAT * EXPN * 128 * 2);   // [b][e][k<=128]
  h16*   wdec  = (h16*)  alloc((size_t)SEQN * 128 * 2);          // decay matrix
  h16*   WiT   = (h16*)  alloc((size_t)HIDN * 128 * 2);
  h16*   inWT  = (h16*)  alloc((size_t)NLAY * 2 * EXPN * HIDN * 2);
  h16*   dtWT  = (h16*)  alloc((size_t)NLAY * EXPN * EXPN * 2);
  h16*   outWT = (h16*)  alloc((size_t)NLAY * HIDN * EXPN * 2);
  float* hbar  = (float*)alloc((size_t)NBAT * HIDN * 4);
  float* hln   = gate; // gate is dead after the last layer; reuse for final LN

  const dim3 blk(256);

  // ---- one-time prep: f16 conversions / weight transposes / decay matrix
  k_cvt<<<(MTOT * 128 + 255) / 256, blk, 0, stream>>>(x, xh, MTOT * 128);
  k_transpose<<<(128 * HIDN + 255) / 256, blk, 0, stream>>>(Wi, WiT, 128, HIDN);
  for (int l = 0; l < NLAY; ++l) {
    k_transpose<<<(HIDN * 2 * EXPN + 255) / 256, blk, 0, stream>>>(
        inW + (size_t)l * HIDN * 2 * EXPN, inWT + (size_t)l * 2 * EXPN * HIDN, HIDN, 2 * EXPN);
    k_transpose<<<(EXPN * EXPN + 255) / 256, blk, 0, stream>>>(
        dtW + (size_t)l * EXPN * EXPN, dtWT + (size_t)l * EXPN * EXPN, EXPN, EXPN);
    k_transpose<<<(EXPN * HIDN + 255) / 256, blk, 0, stream>>>(
        outW + (size_t)l * EXPN * HIDN, outWT + (size_t)l * HIDN * EXPN, EXPN, HIDN);
  }
  k_decay<<<(SEQN * 128 + 255) / 256, blk, 0, stream>>>(wdec);

  // ---- input projection: h = x @ Wi + bi   (M=65536, K=128, N=256)
  gemm_wmma<EP_BIAS><<<dim3(HIDN / 64, MTOT / 128, 1), blk, 0, stream>>>(
      xh, WiT, bi, h, nullptr, nullptr, nullptr, nullptr,
      MTOT, HIDN, 128, 0, 0, 0);

  // ---- 4 mamba blocks
  for (int l = 0; l < NLAY; ++l) {
    k_layernorm<h16><<<MTOT / 8, blk, 0, stream>>>(
        h, ln_w + l * HIDN, ln_b + l * HIDN, hn, MTOT);

    // xz = ln(h) @ inW + inB   (N=1024)
    gemm_wmma<EP_BIAS><<<dim3(2 * EXPN / 64, MTOT / 128, 1), blk, 0, stream>>>(
        hn, inWT + (size_t)l * 2 * EXPN * HIDN, inB + l * 2 * EXPN,
        xz, nullptr, nullptr, nullptr, nullptr,
        MTOT, 2 * EXPN, HIDN, 0, 0, 0);

    // depthwise conv + silu (+ f16 copies for downstream GEMMs)
    k_conv_silu<<<(NBAT * EXPN) / 256, blk, 0, stream>>>(
        xz, convW + (size_t)l * EXPN * 3, convB + l * EXPN, xc, xch, xcT);

    // gate = sigmoid(softplus(xc @ dtW + dtB))   (N=K=512)
    gemm_wmma<EP_GATE><<<dim3(EXPN / 64, MTOT / 128, 1), blk, 0, stream>>>(
        xch, dtWT + (size_t)l * EXPN * EXPN, dtB + l * EXPN,
        gate, nullptr, nullptr, nullptr, nullptr,
        MTOT, EXPN, EXPN, 0, 0, 0);

    // scan: y = (Wdec @ xc[:, :100]) * gate + xc * D, then * silu(z)
    // batched over 32, A shared (stride 0), Bt stride = EXPN*128
    gemm_wmma<EP_SCAN><<<dim3(EXPN / 64, SEQN / 128, NBAT), blk, 0, stream>>>(
        wdec, xcT, Dp + l * EXPN, nullptr, yv, gate, xc, xz,
        SEQN, EXPN, 128, 0, EXPN * 128, SEQN);

    // h += y @ outW + outB   (residual in place)
    gemm_wmma<EP_RESID><<<dim3(HIDN / 64, MTOT / 128, 1), blk, 0, stream>>>(
        yv, outWT + (size_t)l * HIDN * EXPN, outB + l * HIDN,
        h, nullptr, nullptr, nullptr, nullptr,
        MTOT, HIDN, EXPN, 0, 0, 0);
  }

  // ---- head
  k_layernorm<float><<<MTOT / 8, blk, 0, stream>>>(h, fn_w, fn_b, hln, MTOT);
  k_mean<<<NBAT, blk, 0, stream>>>(hln, hbar);
  k_head<<<NBAT, dim3(128), 0, stream>>>(hbar, c1W, c1B, clnW, clnB, c2W, c2B, out);
}
